// Quantize_topk_38362647888335
// MI455X (gfx1250) — matrix operand order repaired and checked by reading, not compiled, not run on hardware
//
#include <hip/hip_runtime.h>
#include <hip/hip_bf16.h>

typedef __attribute__((ext_vector_type(16))) __bf16        v16bf;
typedef __attribute__((ext_vector_type(8)))  float         v8f;
typedef __attribute__((ext_vector_type(8)))  unsigned int  v8u;

#define NROWS 16384              // 16*32*32 query vectors
#define DIM   256
#define NE    4096
#define QOUT  (NROWS * 2 * DIM)  // 8,388,608 quantize_topk elements
#define EPW   (NE / 2)           // codes per wave (2 waves per block)

__device__ __forceinline__ unsigned int pack2_bf16(float a, float b) {
  unsigned int ua = __float_as_uint(a);
  unsigned int ub = __float_as_uint(b);
  // round-to-nearest-even fp32 -> bf16
  ua = ((ua + 0x7FFFu + ((ua >> 16) & 1u)) >> 16) & 0xFFFFu;
  ub = (ub + 0x7FFFu + ((ub >> 16) & 1u)) & 0xFFFF0000u;
  return ua | ub;
}

// order-preserving float -> uint key (ascending): works for all non-NaN
__device__ __forceinline__ unsigned int fkey(float s) {
  unsigned int u = __float_as_uint(s);
  return u ^ (unsigned int)(((int)u >> 31) | 0x80000000);
}

// Transpose codebook [DIM, NE] -> fp32 embT [NE, DIM] (exact gather source),
// bf16 bT [NE, DIM] (WMMA B operand source), and enorm[e] = ||embed[:,e]||^2.
__global__ __launch_bounds__(256) void vq_prep(const float* __restrict__ embed,
                                               float* __restrict__ embT,
                                               unsigned short* __restrict__ bT,
                                               float* __restrict__ enorm) {
  const int e = blockIdx.x;
  const int d = threadIdx.x;
  const float v = embed[(size_t)d * NE + e];
  embT[(size_t)e * DIM + d] = v;
  unsigned int u = __float_as_uint(v);
  bT[(size_t)e * DIM + d] = (unsigned short)((u + 0x7FFFu + ((u >> 16) & 1u)) >> 16);
  __shared__ float red[256];
  red[d] = v * v;
  __syncthreads();
  for (int s = 128; s > 0; s >>= 1) {
    if (d < s) red[d] += red[d + s];
    __syncthreads();
  }
  if (d == 0) enorm[e] = red[0];
}

// 2 waves per 16-row block; each wave scores half of the codebook with bf16
// WMMA (two independent accumulator chains), maintains a branchless packed
// top-2, merges via LDS, then gathers exact fp32 codes + diff partial.
__global__ __launch_bounds__(64) void vq_main(const float* __restrict__ x,
                                              const float* __restrict__ embT,
                                              const unsigned short* __restrict__ bT,
                                              const float* __restrict__ enorm,
                                              float* __restrict__ outq,
                                              float* __restrict__ outind,
                                              float* __restrict__ partial) {
  const int lane = threadIdx.x & 31;
  const int wave = threadIdx.x >> 5;
  const int half = lane >> 4;     // K-half of the fragment this lane holds
  const int col  = lane & 15;     // B column / D column / A row (M) index
  const int rowBase = blockIdx.x * 16;

  __shared__ unsigned long long smW[2][16][2];  // per-wave top-2 per row
  __shared__ unsigned long long smF[16][2];     // merged top-2 per row
  __shared__ float              smD[2];         // per-wave diff partials

  // ---- A fragments: rows rowBase..rowBase+15, fp32 -> bf16, kept in VGPRs ----
  // 16-bit A 16x32 layout: lane M = lane%16; VGPR v: K = (v<4?0:16) + half*8 + 2*(v%4)+{0,1}
  const float* xrow = x + (size_t)(rowBase + col) * DIM;
  v16bf afrag[8];
#pragma unroll
  for (int kt = 0; kt < 8; ++kt) {
    const int b0 = kt * 32 + half * 8;
    v8u au;
#pragma unroll
    for (int i = 0; i < 4; ++i) {
      au[i]     = pack2_bf16(xrow[b0 + 2 * i],      xrow[b0 + 2 * i + 1]);
      au[i + 4] = pack2_bf16(xrow[b0 + 16 + 2 * i], xrow[b0 + 16 + 2 * i + 1]);
    }
    afrag[kt] = __builtin_bit_cast(v16bf, au);
  }

  // Per-lane packed top-2: rows (half*8 + j), columns e ≡ col (mod 16).
  // key = (sortable(score) << 32) | e  -> u64 min gives value AND lowest index.
  unsigned long long b1[8], b2[8];
#pragma unroll
  for (int j = 0; j < 8; ++j) { b1[j] = ~0ull; b2[j] = ~0ull; }

  for (int et = 0; et < EPW / 16; ++et) {
    const int e0 = wave * EPW + et * 16;
    if (et + 1 < EPW / 16)  // WGP-scope prefetch of next B tile
      __builtin_prefetch(bT + (size_t)(e0 + 16 + col) * DIM + half * 16, 0, 3);

    v8f acc0 = {}, acc1 = {};   // two independent WMMA accumulation chains
#pragma unroll
    for (int kt = 0; kt < 8; kt += 2) {
      // 16-bit B 32x16 layout: lane col = lane%16; element s: K = (s<8?0:16) + half*8 + s%8
      const unsigned short* bp0 = bT + (size_t)(e0 + col) * DIM + kt * 32 + half * 8;
      const unsigned short* bp1 = bp0 + 32;
      const uint4 l0 = *(const uint4*)(bp0);
      const uint4 h0 = *(const uint4*)(bp0 + 16);
      const uint4 l1 = *(const uint4*)(bp1);
      const uint4 h1 = *(const uint4*)(bp1 + 16);
      v8u bu0, bu1;
      bu0[0] = l0.x; bu0[1] = l0.y; bu0[2] = l0.z; bu0[3] = l0.w;
      bu0[4] = h0.x; bu0[5] = h0.y; bu0[6] = h0.z; bu0[7] = h0.w;
      bu1[0] = l1.x; bu1[1] = l1.y; bu1[2] = l1.z; bu1[3] = l1.w;
      bu1[4] = h1.x; bu1[5] = h1.y; bu1[6] = h1.z; bu1[7] = h1.w;
      acc0 = __builtin_amdgcn_wmma_f32_16x16x32_bf16(false, afrag[kt], false,
                 __builtin_bit_cast(v16bf, bu0), (short)0, acc0, false, false);
      acc1 = __builtin_amdgcn_wmma_f32_16x16x32_bf16(false, afrag[kt + 1], false,
                 __builtin_bit_cast(v16bf, bu1), (short)0, acc1, false, false);
    }

    // score = ||e||^2 - 2 f.e  (||f||^2 is rank-invariant per row; dropped)
    const float en = enorm[e0 + col];
    const unsigned int ec = (unsigned int)(e0 + col);
#pragma unroll
    for (int j = 0; j < 8; ++j) {
      const float s = __builtin_fmaf(-2.0f, acc0[j] + acc1[j], en);
      const unsigned long long p =
          ((unsigned long long)fkey(s) << 32) | (unsigned long long)ec;
      const unsigned long long lo = (p < b1[j]) ? p : b1[j];   // branchless
      const unsigned long long hi = (p < b1[j]) ? b1[j] : p;
      b1[j] = lo;
      b2[j] = (hi < b2[j]) ? hi : b2[j];
    }
  }

  // ---- merge top-2 across the 16 lanes of each half (butterfly on bits 0..3) ----
#pragma unroll
  for (int m = 1; m < 16; m <<= 1) {
#pragma unroll
    for (int j = 0; j < 8; ++j) {
      const unsigned long long o1 = __shfl_xor(b1[j], m, 32);
      const unsigned long long o2 = __shfl_xor(b2[j], m, 32);
      const unsigned long long n1  = (b1[j] < o1) ? b1[j] : o1;
      const unsigned long long mx  = (b1[j] < o1) ? o1 : b1[j];
      const unsigned long long mn2 = (b2[j] < o2) ? b2[j] : o2;
      b1[j] = n1;
      b2[j] = (mx < mn2) ? mx : mn2;
    }
  }

  // publish per-wave results (one writer lane per row)
#pragma unroll
  for (int j = 0; j < 8; ++j) {
    if (col == j) {
      smW[wave][half * 8 + j][0] = b1[j];
      smW[wave][half * 8 + j][1] = b2[j];
    }
  }
  __syncthreads();

  // wave 0 merges the two waves' sorted pairs and writes final state + indices
  if (wave == 0) {
#pragma unroll
    for (int j = 0; j < 8; ++j) {
      if (col == j) {
        const int r = half * 8 + j;
        const unsigned long long a1 = smW[0][r][0], a2 = smW[0][r][1];
        const unsigned long long c1 = smW[1][r][0], c2 = smW[1][r][1];
        const unsigned long long n1  = (a1 < c1) ? a1 : c1;
        const unsigned long long mx  = (a1 < c1) ? c1 : a1;
        const unsigned long long mn2 = (a2 < c2) ? a2 : c2;
        const unsigned long long n2  = (mx < mn2) ? mx : mn2;
        smF[r][0] = n1;
        smF[r][1] = n2;
        outind[rowBase + r] = (float)(unsigned int)(n1 & 0xFFFFFFFFu);
      }
    }
  }
  __syncthreads();

  // ---- gather exact fp32 codes, write quantize_topk, accumulate diff ----
  // wave 0 -> rows 0..7, wave 1 -> rows 8..15
  float ldiff = 0.0f;
#pragma unroll
  for (int r8 = 0; r8 < 8; ++r8) {
    const int r  = wave * 8 + r8;
    const int e1 = (int)(unsigned int)(smF[r][0] & 0xFFFFFFFFu);
    const int e2 = (int)(unsigned int)(smF[r][1] & 0xFFFFFFFFu);
    const float* c1 = embT + (size_t)e1 * DIM;
    const float* c2 = embT + (size_t)e2 * DIM;
    const float* xr = x + (size_t)(rowBase + r) * DIM;
    float* outr = outq + (size_t)(rowBase + r) * (2 * DIM);
#pragma unroll
    for (int t = 0; t < 8; ++t) {
      const int   d  = t * 32 + lane;
      const float xv = xr[d];
      const float q1 = c1[d];
      const float q2 = c2[d];
      outr[d]       = q1;
      outr[DIM + d] = q2;
      const float d1 = q1 - xv, d2 = q2 - xv;
      ldiff += d1 * d1 + d2 * d2;
    }
  }
#pragma unroll
  for (int m = 1; m < 32; m <<= 1) ldiff += __shfl_xor(ldiff, m, 32);
  if (lane == 0) smD[wave] = ldiff;
  __syncthreads();
  if (threadIdx.x == 0) partial[blockIdx.x] = smD[0] + smD[1];
}

__global__ __launch_bounds__(256) void vq_final(const float* __restrict__ partial,
                                                float* __restrict__ diff) {
  __shared__ float red[256];
  const int t = threadIdx.x;
  red[t] = partial[t] + partial[t + 256] + partial[t + 512] + partial[t + 768];
  __syncthreads();
  for (int s = 128; s > 0; s >>= 1) {
    if (t < s) red[t] += red[t + s];
    __syncthreads();
  }
  if (t == 0) *diff = red[0] / (float)QOUT;
}

extern "C" void kernel_launch(void* const* d_in, const int* in_sizes, int n_in,
                              void* d_out, int out_size, void* d_ws, size_t ws_size,
                              hipStream_t stream) {
  (void)in_sizes; (void)n_in; (void)out_size; (void)ws_size;
  const float* x     = (const float*)d_in[0];   // [16,32,32,256] f32
  const float* embed = (const float*)d_in[1];   // [256,4096] f32

  char* ws = (char*)d_ws;
  float*          embT    = (float*)(ws);                             // 4,194,304 B
  unsigned short* bT      = (unsigned short*)(ws + 4u * 1024 * 1024); // 2,097,152 B
  float*          enorm   = (float*)(ws + 6u * 1024 * 1024);          //    16,384 B
  float*          partial = (float*)(ws + 6u * 1024 * 1024 + 64u * 1024); // 4,096 B

  float* outq   = (float*)d_out;        // [N, 2*DIM]
  float* diff   = outq + QOUT;          // scalar
  float* outind = outq + QOUT + 1;      // [N] nearest-code indices (as float)

  vq_prep <<<NE, 256, 0, stream>>>(embed, embT, bT, enorm);
  vq_main <<<NROWS / 16, 64, 0, stream>>>(x, embT, bT, enorm, outq, outind, partial);
  vq_final<<<1, 256, 0, stream>>>(partial, diff);
}
